// HoughVoting_68367289418399
// MI455X (gfx1250) — compile-verified, matrix operand order checked by reading, not run on hardware
//
#include <hip/hip_runtime.h>
#include <hip/hip_bf16.h>
#include <math.h>

#define NUMC    13
#define HH      120
#define WW      160
#define HW      (HH * WW)      /* 19200 */
#define NCLS    12             /* classes 1..12 */
#define PSAMP   1200           /* HW / 16 sampled voting pixels */
#define PPAD    1216           /* 19 * 64 */
#define NCHUNK  19
#define NTILE   1200           /* HW / 16 center tiles */
#define WPB     4              /* waves per block in voting kernel */

typedef int v8i __attribute__((ext_vector_type(8)));

struct HVStats { int cnt; int ns; float dzsum; float pad; };

// ---------------------------------------------------------------------------
// Kernel 1: per-class stats (num_label, sampled count, sum of log-depth)
// ---------------------------------------------------------------------------
__global__ void hv_stats_kernel(const int* __restrict__ labels,
                                const int* __restrict__ masks,
                                const float* __restrict__ vp,
                                HVStats* __restrict__ stats) {
    const int c   = blockIdx.x + 1;
    const int tid = threadIdx.x;
    const float* dz = vp + (size_t)(3 * c + 2) * HW;
    int cnt = 0, ns = 0;
    float dzs = 0.f;
    for (int i = tid; i < HW; i += 256) {
        bool m = (labels[i] == c) && (masks[i] > 0);
        if (m) {
            cnt++;
            dzs += dz[i];
            if ((i & 15) == 0) ns++;           // sampled pixel (stride 16)
        }
    }
    __shared__ int   scnt[256];
    __shared__ int   sns[256];
    __shared__ float sdz[256];
    scnt[tid] = cnt; sns[tid] = ns; sdz[tid] = dzs;
    __syncthreads();
    for (int off = 128; off > 0; off >>= 1) {
        if (tid < off) {
            scnt[tid] += scnt[tid + off];
            sns[tid]  += sns[tid + off];
            sdz[tid]  += sdz[tid + off];
        }
        __syncthreads();
    }
    if (tid == 0) {
        stats[blockIdx.x].cnt   = scnt[0];
        stats[blockIdx.x].ns    = sns[0];
        stats[blockIdx.x].dzsum = sdz[0];
    }
}

// ---------------------------------------------------------------------------
// Kernel 2: per-class sampled-pixel records {dxn, dyn, xp, yp} (float4).
// Unlabeled / padding pixels get dxn=dyn=0 -> can never be inliers.
// ---------------------------------------------------------------------------
__global__ void hv_pq_kernel(const int* __restrict__ labels,
                             const int* __restrict__ masks,
                             const float* __restrict__ vp,
                             float4* __restrict__ pq4) {
    const int c = blockIdx.x + 1;
    const int s = blockIdx.y * 256 + threadIdx.x;
    if (s >= PPAD) return;
    float dxn = 0.f, dyn = 0.f, xp = 0.f, yp = 0.f;
    if (s < PSAMP) {
        const int i = s * 16;
        xp = (float)((s % 10) * 16);            // W/SKIP = 10 sampled cols
        yp = (float)(s / 10);
        if (labels[i] == c && masks[i] > 0) {
            float dx = vp[(size_t)(3 * c + 0) * HW + i];
            float dy = vp[(size_t)(3 * c + 1) * HW + i];
            float dn = sqrtf(dx * dx + dy * dy) + 1e-6f;
            dxn = dx / dn;
            dyn = dy / dn;
        }
    }
    pq4[(size_t)(c - 1) * PPAD + s] = make_float4(dxn, dyn, xp, yp);
}

// ---------------------------------------------------------------------------
// Kernel 3: Hough voting via IU8 WMMA.
// A = inlier bytes [16 centers x 64 pixels], B = all-ones => C cols = votes.
// Chunk records are DMA'd global->LDS with the CDNA5 async path (ASYNCcnt).
// ---------------------------------------------------------------------------
__global__ void __launch_bounds__(128) hv_vote_kernel(const float4* __restrict__ pq4,
                                                      int* __restrict__ votes) {
    const int cidx     = blockIdx.x;          // class index 0..11
    const int tid      = threadIdx.x;
    const int wave     = tid >> 5;
    const int lane     = tid & 31;
    const int laneHalf = lane >> 4;           // A-matrix K half select
    const int M        = lane & 15;           // A-matrix row (center within tile)
    const int tile     = blockIdx.y * WPB + wave;
    const int g        = tile * 16 + M;       // candidate center pixel index
    const float xg = (float)(g % WW);
    const float yg = (float)(g / WW);

    __shared__ float4 ldsp[64];               // (dxn, dyn, xp, yp) per chunk pixel
    const float4* pqc = pq4 + (size_t)cidx * PPAD;

    // Per-lane constants for the async staging (lanes 0..63 of the block).
    const unsigned ldsOff  = (unsigned)(uintptr_t)(&ldsp[tid]);   // LDS byte offset
    v8i acc = {0, 0, 0, 0, 0, 0, 0, 0};
    v8i bones;
#pragma unroll
    for (int i = 0; i < 8; ++i) bones[i] = 0x01010101;   // B = all ones (u8)

    for (int ch = 0; ch < NCHUNK; ++ch) {
        if (tid < 64) {
            // Async DMA: 16 bytes global -> LDS per lane, tracked by ASYNCcnt.
            unsigned long long gaddr =
                (unsigned long long)(const void*)(&pqc[ch * 64 + tid]);
            asm volatile("global_load_async_to_lds_b128 %0, %1, off"
                         :: "v"(ldsOff), "v"(gaddr) : "memory");
            asm volatile("s_wait_asynccnt 0x0" ::: "memory");
        }
        if (ch + 1 < NCHUNK)                              // warm next chunk in L2
            __builtin_prefetch(&pqc[(ch + 1) * 64 + (tid & 63)], 0, 1);
        __syncthreads();

        v8i a;
#pragma unroll
        for (int v = 0; v < 8; ++v) {
            int aw = 0;
#pragma unroll
            for (int b = 0; b < 4; ++b) {
                // ISA 8-bit A 16x64 layout: byte b of dword v holds K below
                const int K = (v >> 1) * 16 + (v & 1) * 4 + b + laneHalf * 8;
                float4 q = ldsp[K];
                float ddx   = xg - q.z;
                float ddy   = yg - q.w;
                float dist2 = ddx * ddx + ddy * ddy;   // exact small integer
                float dot   = ddx * q.x + ddy * q.y;
                // cos>0.9 && dist>0  <=>  dot>0 && dot^2 > 0.81*dist^2
                int in = ((dot > 0.0f) && (dot * dot > 0.81f * dist2)) ? 1 : 0;
                aw |= in << (8 * b);
            }
            a[v] = aw;
        }
        __syncthreads();
        acc = __builtin_amdgcn_wmma_i32_16x16x64_iu8(false, a, false, bones, acc,
                                                     false, false);
    }

    // C/D layout: lanes 0-15 vgpr r -> M=r ; lanes 16-31 vgpr r -> M=8+r.
    int* vout = votes + (size_t)cidx * HW + tile * 16;
    if (lane == 0) {
#pragma unroll
        for (int r = 0; r < 8; ++r) vout[r] = acc[r];
    } else if (lane == 16) {
#pragma unroll
        for (int r = 0; r < 8; ++r) vout[8 + r] = acc[r];
    }
}

// ---------------------------------------------------------------------------
// Kernel 4: argmax (first-max tie-break, like jnp.argmax) + output assembly.
// ---------------------------------------------------------------------------
__global__ void hv_final_kernel(const int* __restrict__ votes,
                                const HVStats* __restrict__ stats,
                                const float* __restrict__ extents,
                                const float* __restrict__ meta,
                                float* __restrict__ out) {
    const int cidx = blockIdx.x;
    const int c    = cidx + 1;
    const int tid  = threadIdx.x;
    const int* v   = votes + (size_t)cidx * HW;

    int bv = -1, bi = 0;
    for (int i = tid; i < HW; i += 256) {     // strided ascending -> strict >
        int x = v[i];
        if (x > bv) { bv = x; bi = i; }
    }
    __shared__ int sv[256];
    __shared__ int si[256];
    sv[tid] = bv; si[tid] = bi;
    __syncthreads();
    for (int off = 128; off > 0; off >>= 1) {
        if (tid < off) {
            int v2 = sv[tid + off], i2 = si[tid + off];
            if (v2 > sv[tid] || (v2 == sv[tid] && i2 < si[tid])) {
                sv[tid] = v2; si[tid] = i2;
            }
        }
        __syncthreads();
    }

    __shared__ float validf;
    if (tid == 0) {
        const int   best = si[0];
        const float vmax = (float)sv[0];
        const float cx = (float)(best % WW);
        const float cy = (float)(best / WW);
        HVStats st = stats[cidx];
        float msum = fmaxf((float)st.cnt, 1.0f);
        float z    = fmaxf(expf(st.dzsum / msum), 1e-3f);
        float fx = meta[0], fy = meta[4], pxc = meta[2], pyc = meta[5];
        float bw = extents[c * 3 + 0] * fx / z;
        float bh = extents[c * 3 + 1] * fy / z;
        float nsamp = fmaxf((float)st.ns, 1.0f);
        bool valid = (vmax >= 10.0f) && (st.cnt >= 500) && (vmax >= 0.02f * nsamp);

        float* box = out + cidx * 7;                       // top_box [12,7]
        box[0] = 0.0f;                                     // batch index
        box[1] = (float)c;
        box[2] = cx - bw * 0.5f;
        box[3] = cy - bh * 0.5f;
        box[4] = cx + bw * 0.5f;
        box[5] = cy + bh * 0.5f;
        box[6] = vmax;

        float* pose = out + 84 + cidx * 7;                 // top_pose [12,7]
        pose[0] = 1.0f; pose[1] = 0.0f; pose[2] = 0.0f; pose[3] = 0.0f;
        pose[4] = (cx - pxc) * z / fx;
        pose[5] = (cy - pyc) * z / fy;
        pose[6] = z;

        out[1416 + cidx] = 0.0f;                           // top_domain (int32 0 bits)
        validf = valid ? 1.0f : 0.0f;
    }
    __syncthreads();
    if (tid < 52) {
        out[168 + cidx * 52 + tid] = 0.0f;                 // top_target
        out[792 + cidx * 52 + tid] = ((tid >> 2) == c) ? validf : 0.0f;  // top_weight
    }
}

// ---------------------------------------------------------------------------
extern "C" void kernel_launch(void* const* d_in, const int* in_sizes, int n_in,
                              void* d_out, int out_size, void* d_ws, size_t ws_size,
                              hipStream_t stream) {
    const int*   labels  = (const int*)d_in[0];
    const int*   masks   = (const int*)d_in[1];
    const float* vp      = (const float*)d_in[2];
    const float* extents = (const float*)d_in[3];
    /* d_in[4] = poses (unused by forward outputs) */
    const float* meta    = (const float*)d_in[5];
    float*       out     = (float*)d_out;

    char* ws = (char*)d_ws;
    float4*  pq4   = (float4*)ws;                                 // 12*1216 float4
    size_t   off   = (size_t)NCLS * PPAD * sizeof(float4);        // 233472 B
    int*     votes = (int*)(ws + off);                            // 12*19200 i32
    off           += (size_t)NCLS * HW * sizeof(int);             // +921600 B
    HVStats* stats = (HVStats*)(ws + off);                        // 12 structs

    hv_stats_kernel<<<NCLS, 256, 0, stream>>>(labels, masks, vp, stats);
    hv_pq_kernel<<<dim3(NCLS, (PPAD + 255) / 256), 256, 0, stream>>>(labels, masks,
                                                                     vp, pq4);
    hv_vote_kernel<<<dim3(NCLS, NTILE / WPB), 128, 0, stream>>>(pq4, votes);
    hv_final_kernel<<<NCLS, 256, 0, stream>>>(votes, stats, extents, meta, out);
}
